// SelfAttn_46737834115264
// MI455X (gfx1250) — matrix operand order repaired and checked by reading, not compiled, not run on hardware
//
#include <hip/hip_runtime.h>
#include <hip/hip_bf16.h>

// ---------------------------------------------------------------------------
// Problem constants (match reference): B=2, S=2048, D=1024, H=16, E=64
// ---------------------------------------------------------------------------
constexpr int Bc = 2;
constexpr int Sc = 2048;
constexpr int Dc = 1024;
constexpr int Hc = 16;
constexpr int Ec = 64;

typedef __attribute__((ext_vector_type(16))) __bf16 v16bf;
typedef __attribute__((ext_vector_type(8)))  __bf16 v8bf;
typedef __attribute__((ext_vector_type(8)))  float  v8f;

#define WMMA_BF16(a, b, c) \
    __builtin_amdgcn_wmma_f32_16x16x32_bf16(false, (a), false, (b), (short)0, (c), false, false)

union FragAB {
    v16bf v;
    v8bf  h[2];
};

__device__ __forceinline__ v8f zero8() {
    v8f z = {0.f, 0.f, 0.f, 0.f, 0.f, 0.f, 0.f, 0.f};
    return z;
}

// A fragment (16x32 bf16, row-major source, lda in elements).
// Lane holds row = lane&15; elements [0..15] = K k0+half*16+i  (half = lane>>4).
__device__ __forceinline__ v16bf load_a_frag(const __bf16* A, int lda, int row0, int k0, int lane) {
    FragAB f;
    const __bf16* p = A + (size_t)(row0 + (lane & 15)) * lda + k0 + ((lane >> 4) << 4);
    f.h[0] = *(const v8bf*)p;
    f.h[1] = *(const v8bf*)(p + 8);
    return f.v;
}

// B fragment from 16 contiguous bf16 (per-lane): col = lane&15,
// K = (lane>>4)*16 + idx. Caller computes the per-lane base pointer.
__device__ __forceinline__ v16bf load_b_frag16(const __bf16* p) {
    FragAB f;
    f.h[0] = *(const v8bf*)p;
    f.h[1] = *(const v8bf*)(p + 8);
    return f.v;
}

// Load one k-step's fragment set: 2 A-frags (32 rows) + 4 B-frags (64 cols).
__device__ __forceinline__ void gemm_load(const __bf16* A, int lda, const __bf16* Wb,
                                          int m0, int kb, int lane,
                                          v16bf a[2], v16bf b[4]) {
    a[0] = load_a_frag(A, lda, m0, kb * 32, lane);
    a[1] = load_a_frag(A, lda, m0 + 16, kb * 32, lane);
#pragma unroll
    for (int nt = 0; nt < 4; ++nt)
        b[nt] = load_b_frag16(Wb + ((((size_t)kb * 4 + nt) * 32 + lane) << 4));
}

// 8 WMMAs consuming one fragment set.
__device__ __forceinline__ void gemm_mma(v8f acc[2][4], const v16bf a[2], const v16bf b[4]) {
#pragma unroll
    for (int mt = 0; mt < 2; ++mt)
#pragma unroll
        for (int nt = 0; nt < 4; ++nt)
            acc[mt][nt] = WMMA_BF16(a[mt], b[nt], acc[mt][nt]);
}

// ---------------------------------------------------------------------------
// Elementwise fp32 -> bf16 convert
// ---------------------------------------------------------------------------
__global__ void cvt_bf16_kernel(const float* __restrict__ in, __bf16* __restrict__ out, long n) {
    long i = (long)blockIdx.x * blockDim.x + threadIdx.x;
    if (i < n) out[i] = (__bf16)in[i];
}

// ---------------------------------------------------------------------------
// Weight swizzle: [G][Dk][Ek] fp32 row-major  ->  bf16 in B-fragment order:
// idx = (((g*(Dk/32)+kb)*(Ek/16)+nt)*32 + lane)*16 + kk
// value = W[g][kb*32 + (lane>>4)*16 + kk][nt*16 + (lane&15)]
// so each lane's B fragment is one contiguous 32-byte read.
// ---------------------------------------------------------------------------
__global__ void swz_w_kernel(const float* __restrict__ w, __bf16* __restrict__ out,
                             int G, int Dk, int Ek) {
    long i = (long)blockIdx.x * blockDim.x + threadIdx.x;
    long total = (long)G * Dk * Ek;
    if (i >= total) return;
    int kk = (int)(i & 15);
    long t = i >> 4;
    int lane = (int)(t & 31);
    t >>= 5;
    int ntc = Ek >> 4, kbc = Dk >> 5;
    int nt = (int)(t % ntc);
    t /= ntc;
    int kb = (int)(t % kbc);
    int g  = (int)(t / kbc);
    int krow = kb * 32 + ((lane >> 4) << 4) + kk;
    int col  = nt * 16 + (lane & 15);
    out[i] = (__bf16)w[((long)g * Dk + krow) * Ek + col];
}

// ---------------------------------------------------------------------------
// QKV projection: for each (b, sel, h):  O = X[b] (SxD) @ Wswz[sel][h] (DxE)
// grid = (S/256, 3*H, B), block = 256 (8 waves).
// Wave owns a 32x64 output tile. The k-loop is 2-deep modulo-scheduled with
// alternating named buffers -> no register rotation copies in steady state.
// ---------------------------------------------------------------------------
__global__ __launch_bounds__(256) void qkv_gemm_kernel(
    const __bf16* __restrict__ xbf,
    const __bf16* __restrict__ wq, const __bf16* __restrict__ wk, const __bf16* __restrict__ wv,
    __bf16* __restrict__ q, __bf16* __restrict__ k, __bf16* __restrict__ v) {
    const int lane = threadIdx.x & 31;
    const int wave = threadIdx.x >> 5;
    const int sel  = blockIdx.y / Hc;
    const int hh   = blockIdx.y % Hc;
    const int bb   = blockIdx.z;
    const int m0   = blockIdx.x * 256 + wave * 32;

    const __bf16* wsel = (sel == 0) ? wq : (sel == 1) ? wk : wv;
    __bf16*       osel = (sel == 0) ? q  : (sel == 1) ? k  : v;

    const __bf16* A  = xbf + (size_t)bb * Sc * Dc;
    const __bf16* Wb = wsel + (size_t)hh * Dc * Ec;         // swizzled per-head block
    __bf16*       O  = osel + ((size_t)bb * Hc + hh) * Sc * Ec;

    v8f acc[2][4];
#pragma unroll
    for (int mt = 0; mt < 2; ++mt)
#pragma unroll
        for (int nt = 0; nt < 4; ++nt) acc[mt][nt] = zero8();

    constexpr int KB = Dc / 32;  // 32 k-steps (even)
    v16bf aA[2], bA[4], aB[2], bB[4];
    gemm_load(A, Dc, Wb, m0, 0, lane, aA, bA);
    gemm_load(A, Dc, Wb, m0, 1, lane, aB, bB);

    int kb = 0;
    for (; kb + 2 < KB; kb += 2) {
        gemm_mma(acc, aA, bA);
        gemm_load(A, Dc, Wb, m0, kb + 2, lane, aA, bA);   // in flight over next 8 WMMAs
        gemm_mma(acc, aB, bB);
        gemm_load(A, Dc, Wb, m0, kb + 3, lane, aB, bB);
    }
    gemm_mma(acc, aA, bA);
    gemm_mma(acc, aB, bB);

    // C/D layout: row = m0 + mt*16 + (lane>>4)*8 + r, col = nt*16 + (lane&15)
    const int cbase = lane & 15;
#pragma unroll
    for (int mt = 0; mt < 2; ++mt) {
        const int rbase = m0 + mt * 16 + ((lane >> 4) << 3);
#pragma unroll
        for (int nt = 0; nt < 4; ++nt)
#pragma unroll
            for (int r = 0; r < 8; ++r)
                O[(size_t)(rbase + r) * Ec + nt * 16 + cbase] = (__bf16)acc[mt][nt][r];
    }
}

// ---------------------------------------------------------------------------
// Flash attention (causal) per (b,h).
// grid = (S/128, H, B), block = 256 (8 waves). Wave owns 16 query rows.
// Key blocks of 32 staged in LDS; next tile's global loads are prefetched
// into registers during current-tile compute. Online softmax; P transposed
// through per-wave LDS (C layout -> A layout).
// ---------------------------------------------------------------------------
__global__ __launch_bounds__(256) void attn_kernel(
    const __bf16* __restrict__ Q, const __bf16* __restrict__ K,
    const __bf16* __restrict__ V, __bf16* __restrict__ ctx) {
    __shared__ __bf16 Ksh[32][72];       // key tile, row-major [t][e], padded to 16B mult.
    __shared__ __bf16 Vsh[64][40];       // value tile TRANSPOSED [e][t], padded
    __shared__ __bf16 Psh[8][16][32];    // per-wave probs for C->A layout transpose

    const int lane = threadIdx.x & 31;
    const int wave = threadIdx.x >> 5;
    const int hh   = blockIdx.y;
    const int bb   = blockIdx.z;
    const int q0w  = blockIdx.x * 128 + wave * 16;

    const size_t off = ((size_t)bb * Hc + hh) * Sc * Ec;
    const __bf16* Qh = Q + off;
    const __bf16* Kh = K + off;
    const __bf16* Vh = V + off;

    // Q fragments for this wave's 16 rows (K-dim = E = 64 -> 2 fragments), kept in regs.
    v16bf qa[2];
    qa[0] = load_a_frag(Qh, Ec, q0w, 0, lane);
    qa[1] = load_a_frag(Qh, Ec, q0w, 32, lane);

    v8f o[4];
#pragma unroll
    for (int i = 0; i < 4; ++i) o[i] = zero8();
    float m[8], l[8];
#pragma unroll
    for (int r = 0; r < 8; ++r) { m[r] = -3.0e38f; l[r] = 0.0f; }

    const float LOG2E = 1.44269504088896340736f;
    const float scale = 0.125f;  // 1/sqrt(64)

    // Tile loader indices: 256 threads cover 32 rows x 8 segments of 8 bf16.
    const int lt  = threadIdx.x >> 3;   // 0..31 : tile row (key index in block)
    const int seg = threadIdx.x & 7;    // 0..7  : 8-element column segment

    const int nkb = blockIdx.x * 4 + 4;  // causal: only key blocks <= query block

    // Prefetch tile 0 into registers.
    v8bf kreg = *(const v8bf*)(Kh + (size_t)lt * Ec + seg * 8);
    v8bf vreg = *(const v8bf*)(Vh + (size_t)lt * Ec + seg * 8);

    for (int kb = 0; kb < nkb; ++kb) {
        const int colbase = kb * 32;
        __syncthreads();  // previous tile fully consumed
        // Stage prefetched K (row-major) and V (transposed) tiles into LDS.
        *(v8bf*)&Ksh[lt][seg * 8] = kreg;
#pragma unroll
        for (int j = 0; j < 8; ++j) Vsh[seg * 8 + j][lt] = vreg[j];
        __syncthreads();  // tiles ready

        // Prefetch next tile from global; loads overlap with compute below.
        if (kb + 1 < nkb) {
            const size_t nb = (size_t)(colbase + 32 + lt) * Ec + seg * 8;
            kreg = *(const v8bf*)(Kh + nb);
            vreg = *(const v8bf*)(Vh + nb);
        }

        const bool active = colbase <= (q0w + 15);  // wave-uniform
        if (active) {
            // ---- scores: S = Q (16xE) @ K^T (Ex32), 2 N-tiles x 2 K-steps ----
            v8f sc[2];
            sc[0] = zero8();
            sc[1] = zero8();
#pragma unroll
            for (int nt = 0; nt < 2; ++nt)
#pragma unroll
                for (int es = 0; es < 2; ++es) {
                    const __bf16* kp = &Ksh[nt * 16 + (lane & 15)][es * 32 + ((lane >> 4) << 4)];
                    sc[nt] = WMMA_BF16(qa[es], load_b_frag16(kp), sc[nt]);
                }

            // ---- scale + causal mask (C layout) ----
#pragma unroll
            for (int nt = 0; nt < 2; ++nt)
#pragma unroll
                for (int r = 0; r < 8; ++r) {
                    int col = colbase + nt * 16 + (lane & 15);
                    int row = q0w + ((lane >> 4) << 3) + r;
                    float sv = sc[nt][r] * scale;
                    sc[nt][r] = (col <= row) ? sv : -3.0e38f;
                }

            // ---- online softmax: per-row reductions across 16-lane halves ----
#pragma unroll
            for (int r = 0; r < 8; ++r) {
                float mx = fmaxf(sc[0][r], sc[1][r]);
                mx = fmaxf(mx, __shfl_xor(mx, 1, 16));
                mx = fmaxf(mx, __shfl_xor(mx, 2, 16));
                mx = fmaxf(mx, __shfl_xor(mx, 4, 16));
                mx = fmaxf(mx, __shfl_xor(mx, 8, 16));
                float mnew  = fmaxf(m[r], mx);
                float alpha = exp2f((m[r] - mnew) * LOG2E);
                float p0 = exp2f((sc[0][r] - mnew) * LOG2E);
                float p1 = exp2f((sc[1][r] - mnew) * LOG2E);
                sc[0][r] = p0;
                sc[1][r] = p1;
                float rs = p0 + p1;
                rs += __shfl_xor(rs, 1, 16);
                rs += __shfl_xor(rs, 2, 16);
                rs += __shfl_xor(rs, 4, 16);
                rs += __shfl_xor(rs, 8, 16);
                l[r] = l[r] * alpha + rs;
                m[r] = mnew;
                o[0][r] *= alpha;
                o[1][r] *= alpha;
                o[2][r] *= alpha;
                o[3][r] *= alpha;
            }

            // ---- transpose P (C layout -> A layout) through per-wave LDS ----
#pragma unroll
            for (int nt = 0; nt < 2; ++nt)
#pragma unroll
                for (int r = 0; r < 8; ++r)
                    Psh[wave][((lane >> 4) << 3) + r][nt * 16 + (lane & 15)] = (__bf16)sc[nt][r];
            asm volatile("s_wait_dscnt 0x0" ::: "memory");  // intra-wave LDS RAW

            FragAB pf;
            const __bf16* pp = &Psh[wave][lane & 15][(lane >> 4) << 4];
            pf.h[0] = *(const v8bf*)pp;
            pf.h[1] = *(const v8bf*)(pp + 8);

            // ---- O += P (16x32) @ V (32xE); B operand contiguous from Vsh^T ----
#pragma unroll
            for (int nt = 0; nt < 4; ++nt) {
                const __bf16* vp = &Vsh[nt * 16 + (lane & 15)][(lane >> 4) << 4];
                o[nt] = WMMA_BF16(pf.v, load_b_frag16(vp), o[nt]);
            }
        }
    }

    // ---- normalize and write ctx in head-concat layout [B][S][H*E] ----
#pragma unroll
    for (int r = 0; r < 8; ++r) {
        float inv = 1.0f / l[r];
        int row = q0w + ((lane >> 4) << 3) + r;
#pragma unroll
        for (int nt = 0; nt < 4; ++nt) {
            int col = nt * 16 + (lane & 15);
            ctx[((size_t)bb * Sc + row) * (Hc * Ec) + hh * Ec + col] = (__bf16)(o[nt][r] * inv);
        }
    }
}

// ---------------------------------------------------------------------------
// Output projection: [B*S, H*E] bf16 @ Wswz (H*E x E) -> fp32 out [B*S, E]
// grid = (B*S/256, 1, 1), block = 256. Wave owns 32x64; 2-deep modulo-
// scheduled k-loop with alternating buffers (no rotation copies).
// ---------------------------------------------------------------------------
__global__ __launch_bounds__(256) void out_gemm_kernel(
    const __bf16* __restrict__ ctx, const __bf16* __restrict__ wswz,
    float* __restrict__ out) {
    const int lane = threadIdx.x & 31;
    const int wave = threadIdx.x >> 5;
    const int m0   = blockIdx.x * 256 + wave * 32;
    constexpr int Kd = Hc * Ec;  // 1024
    constexpr int KB = Kd / 32;  // 32 (even)

    v8f acc[2][4];
#pragma unroll
    for (int mt = 0; mt < 2; ++mt)
#pragma unroll
        for (int nt = 0; nt < 4; ++nt) acc[mt][nt] = zero8();

    v16bf aA[2], bA[4], aB[2], bB[4];
    gemm_load(ctx, Kd, wswz, m0, 0, lane, aA, bA);
    gemm_load(ctx, Kd, wswz, m0, 1, lane, aB, bB);

    int kb = 0;
    for (; kb + 2 < KB; kb += 2) {
        gemm_mma(acc, aA, bA);
        gemm_load(ctx, Kd, wswz, m0, kb + 2, lane, aA, bA);
        gemm_mma(acc, aB, bB);
        gemm_load(ctx, Kd, wswz, m0, kb + 3, lane, aB, bB);
    }
    gemm_mma(acc, aA, bA);
    gemm_mma(acc, aB, bB);

    const int cbase = lane & 15;
#pragma unroll
    for (int mt = 0; mt < 2; ++mt) {
        const int rbase = m0 + mt * 16 + ((lane >> 4) << 3);
#pragma unroll
        for (int nt = 0; nt < 4; ++nt)
#pragma unroll
            for (int r = 0; r < 8; ++r)
                out[(size_t)(rbase + r) * Ec + nt * 16 + cbase] = acc[mt][nt][r];
    }
}

// ---------------------------------------------------------------------------
// Launch
// ---------------------------------------------------------------------------
extern "C" void kernel_launch(void* const* d_in, const int* in_sizes, int n_in,
                              void* d_out, int out_size, void* d_ws, size_t ws_size,
                              hipStream_t stream) {
    (void)in_sizes; (void)n_in; (void)out_size; (void)ws_size;

    const float* x  = (const float*)d_in[0];
    const float* Wq = (const float*)d_in[1];
    const float* Wk = (const float*)d_in[2];
    const float* Wv = (const float*)d_in[3];
    const float* W  = (const float*)d_in[4];

    // Workspace carve-up (~46 MB total)
    const size_t n_x   = (size_t)Bc * Sc * Dc;
    const size_t n_w   = (size_t)Hc * Dc * Ec;
    const size_t n_wo  = (size_t)(Hc * Ec) * Ec;
    const size_t n_qkv = (size_t)Bc * Hc * Sc * Ec;
    const size_t n_ctx = (size_t)Bc * Sc * Hc * Ec;

    char* p = (char*)d_ws;
    __bf16* xbf  = (__bf16*)p; p += n_x * 2;
    __bf16* wqs  = (__bf16*)p; p += n_w * 2;
    __bf16* wks  = (__bf16*)p; p += n_w * 2;
    __bf16* wvs  = (__bf16*)p; p += n_w * 2;
    __bf16* wos  = (__bf16*)p; p += n_wo * 2;
    __bf16* qb   = (__bf16*)p; p += n_qkv * 2;
    __bf16* kb   = (__bf16*)p; p += n_qkv * 2;
    __bf16* vb   = (__bf16*)p; p += n_qkv * 2;
    __bf16* ctxb = (__bf16*)p; p += n_ctx * 2;

    cvt_bf16_kernel<<<(unsigned)((n_x + 255) / 256), 256, 0, stream>>>(x, xbf, (long)n_x);
    swz_w_kernel<<<(unsigned)((n_w + 255) / 256), 256, 0, stream>>>(Wq, wqs, Hc, Dc, Ec);
    swz_w_kernel<<<(unsigned)((n_w + 255) / 256), 256, 0, stream>>>(Wk, wks, Hc, Dc, Ec);
    swz_w_kernel<<<(unsigned)((n_w + 255) / 256), 256, 0, stream>>>(Wv, wvs, Hc, Dc, Ec);
    swz_w_kernel<<<(unsigned)((n_wo + 255) / 256), 256, 0, stream>>>(W, wos, 1, Hc * Ec, Ec);

    qkv_gemm_kernel<<<dim3(Sc / 256, 3 * Hc, Bc), 256, 0, stream>>>(xbf, wqs, wks, wvs, qb, kb, vb);
    attn_kernel<<<dim3(Sc / 128, Hc, Bc), 256, 0, stream>>>(qb, kb, vb, ctxb);
    out_gemm_kernel<<<dim3((Bc * Sc) / 256, 1, 1), 256, 0, stream>>>(ctxb, wos, (float*)d_out);
}
